// MultiHeadSelfAttentionModern_33457795236735
// MI455X (gfx1250) — matrix operand order repaired and checked by reading, not compiled, not run on hardware
//
#include <hip/hip_runtime.h>
#include <hip/hip_bf16.h>

// ---------------- types ----------------
typedef __attribute__((ext_vector_type(16))) __bf16 v16bf;
typedef __attribute__((ext_vector_type(8)))  float  v8f;
typedef unsigned short bfu; // raw bf16 bits

union FragU { v16bf v; uint4 u4[2]; };

// ---------------- problem constants ----------------
#define T_LEN    2048
#define D_MODEL  2048
#define DH       64
#define NH       32
#define NKV      8
#define SINKN    64
#define S_ALL    2112      // SINK + WINDOW = 64 + 2048 (33 tiles of 64)
#define START_POS 2048

__device__ __forceinline__ bfu f2bf(float f) {
    union { float f; unsigned u; } v; v.f = f;
    unsigned r = v.u + 0x7FFFu + ((v.u >> 16) & 1u); // RNE
    return (bfu)(r >> 16);
}
__device__ __forceinline__ unsigned pack2(float lo, float hi) {
    return (unsigned)f2bf(lo) | ((unsigned)f2bf(hi) << 16);
}

// Async global->LDS copy of 64 bytes (4 x B128), ASYNCcnt-tracked (CDNA5 TDM-lite path).
__device__ __forceinline__ void async_copy64(const void* gsrc, void* ldst) {
    unsigned lds = (unsigned)(uintptr_t)ldst;                 // low 32 bits = LDS offset
    unsigned long long ga = (unsigned long long)(uintptr_t)gsrc;
    asm volatile(
        "global_load_async_to_lds_b128 %0, %1, off\n\t"
        "global_load_async_to_lds_b128 %0, %1, off offset:16\n\t"
        "global_load_async_to_lds_b128 %0, %1, off offset:32\n\t"
        "global_load_async_to_lds_b128 %0, %1, off offset:48"
        :: "v"(lds), "v"(ga) : "memory");
}
__device__ __forceinline__ void async_copy32(const void* gsrc, void* ldst) {
    unsigned lds = (unsigned)(uintptr_t)ldst;
    unsigned long long ga = (unsigned long long)(uintptr_t)gsrc;
    asm volatile(
        "global_load_async_to_lds_b128 %0, %1, off\n\t"
        "global_load_async_to_lds_b128 %0, %1, off offset:16"
        :: "v"(lds), "v"(ga) : "memory");
}
__device__ __forceinline__ void wait_async0() {
    asm volatile("s_wait_asynccnt 0x0" ::: "memory");
}

// ---------------- kernel 1: f32 -> bf16 (x) ----------------
__global__ void cvt_f32_bf16(const float* __restrict__ src, bfu* __restrict__ dst) {
    int i = blockIdx.x * blockDim.x + threadIdx.x; // one float4 per thread
    float4 f = ((const float4*)src)[i];
    uint2 pk; pk.x = pack2(f.x, f.y); pk.y = pack2(f.z, f.w);
    ((uint2*)dst)[i] = pk;
}

// ---------------- kernel 2: cache sink -> K/V buffers ----------------
__global__ void sink_copy(const float* __restrict__ cache,
                          bfu* __restrict__ kall, bfu* __restrict__ vall) {
    int i = (blockIdx.x * blockDim.x + threadIdx.x) * 4; // NKV*SINKN*DH = 32768 elems
    int kvh = i / (SINKN * DH);
    int rem = i - kvh * (SINKN * DH);                    // s*DH + d, s in [0,64)
    float4 f = *(const float4*)(cache + (size_t)kvh * (2048 * DH) + rem);
    uint2 pk; pk.x = pack2(f.x, f.y); pk.y = pack2(f.z, f.w);
    *(uint2*)(kall + (size_t)kvh * (S_ALL * DH) + rem) = pk;
    *(uint2*)(vall + (size_t)kvh * (S_ALL * DH) + rem) = pk;
}

// ---------------- GEMM with fused epilogues ----------------
// C[M,N] = A[M,K](bf16) x B[K,N](f32 in global, converted to bf16 in LDS)
// BM=128 BN=64 BK=32 ; 256 threads = 8 waves ; wave w -> rows [16w,16w+16), all 64 cols
enum { MODE_Q = 0, MODE_K = 1, MODE_V = 2, MODE_O = 3 };

template <int MODE>
__global__ __launch_bounds__(256)
void gemm_bf16(const bfu* __restrict__ A, const float* __restrict__ B,
               bfu* __restrict__ outb, float* __restrict__ outf,
               const float* __restrict__ bias, int N) {
    const int K = D_MODEL;
    __shared__ bfu As[128 * 40];   // [row][k] pad->40
    __shared__ bfu Bs[64 * 40];    // transposed: [n][k] pad->40

    const int tid  = threadIdx.x;
    const int wave = tid >> 5, lane = tid & 31;
    const int half = lane >> 4, l16 = lane & 15;
    const int bm = blockIdx.x, bn = blockIdx.y;

    const bfu*   Ap = A + (size_t)(bm * 128) * K;
    const float* Bp = B + (size_t)(bn * 64);

    v8f acc[4];
#pragma unroll
    for (int i = 0; i < 4; ++i)
#pragma unroll
        for (int j = 0; j < 8; ++j) acc[i][j] = 0.0f;

    const int arow = tid >> 1, aseg = tid & 1;      // A stage: 32B each (async)
    const int kp = (tid >> 4) << 1;                 // B stage: k-pair 0..30
    const int n0 = (tid & 15) * 4;                  // B stage: 4 cols

    for (int kt = 0; kt < K / 32; ++kt) {
        // stage A tile 128x32 bf16 : async global->LDS, no VGPR round-trip
        async_copy32(Ap + (size_t)arow * K + kt * 32 + aseg * 16,
                     As + arow * 40 + aseg * 16);
        { // stage B tile 32x64 f32 -> bf16, transposed, packed pair stores
            const float* g = Bp + (size_t)(kt * 32 + kp) * N + n0;
            float4 f0 = ((const float4*)g)[0];
            float4 f1 = ((const float4*)(g + N))[0];
            *(unsigned*)(Bs + (n0 + 0) * 40 + kp) = pack2(f0.x, f1.x);
            *(unsigned*)(Bs + (n0 + 1) * 40 + kp) = pack2(f0.y, f1.y);
            *(unsigned*)(Bs + (n0 + 2) * 40 + kp) = pack2(f0.z, f1.z);
            *(unsigned*)(Bs + (n0 + 3) * 40 + kp) = pack2(f0.w, f1.w);
        }
        if (kt + 1 < K / 32) { // prefetch next B panel -> global_prefetch_b8
            __builtin_prefetch(Bp + (size_t)((kt + 1) * 32 + kp) * N + n0, 0, 1);
        }
        wait_async0();
        __syncthreads();

        FragU a;
        { // A fragment: row = 16*wave + l16 ; K = 16*half + e
            const bfu* ap = As + (wave * 16 + l16) * 40;
            a.u4[0] = *(const uint4*)(ap + 16 * half);
            a.u4[1] = *(const uint4*)(ap + 16 * half + 8);
        }
#pragma unroll
        for (int nt = 0; nt < 4; ++nt) {
            FragU b; // B fragment: col = 16*nt + l16 ; K = 16*half + e
            const bfu* bp = Bs + (nt * 16 + l16) * 40 + 16 * half;
            b.u4[0] = ((const uint4*)bp)[0];
            b.u4[1] = ((const uint4*)bp)[1];
            acc[nt] = __builtin_amdgcn_wmma_f32_16x16x32_bf16(
                false, a.v, false, b.v, (short)0, acc[nt], false, false);
        }
        __syncthreads();
    }

    // ---- epilogue ---- (C frag: row = r + 8*half, col = 16*nt + l16)
    const int row_base = bm * 128 + wave * 16 + 8 * half;
    if (MODE == MODE_O) {
#pragma unroll
        for (int nt = 0; nt < 4; ++nt) {
            int col = bn * 64 + nt * 16 + l16;
            float bo = bias[col];
#pragma unroll
            for (int r = 0; r < 8; ++r)
                outf[(size_t)(row_base + r) * N + col] = acc[nt][r] + bo;
        }
    } else if (MODE == MODE_V) {
#pragma unroll
        for (int nt = 0; nt < 4; ++nt) {
            int d = nt * 16 + l16;
#pragma unroll
            for (int r = 0; r < 8; ++r)
                outb[((size_t)bn * S_ALL + SINKN + row_base + r) * DH + d] = f2bf(acc[nt][r]);
        }
    } else { // Q or K: fused RoPE (pairs d, d+32) ; Q also folds 1/sqrt(DH)
#pragma unroll
        for (int nt = 0; nt < 2; ++nt) {
            int d = nt * 16 + l16;                              // 0..31
            float inv = expf(-0.28782313662425572f * (float)d); // 10000^(-2d/64)
#pragma unroll
            for (int r = 0; r < 8; ++r) {
                int row = row_base + r;
                float ang = (float)(START_POS + row) * inv;
                float s, c; sincosf(ang, &s, &c);
                float x1 = acc[nt][r], x2 = acc[nt + 2][r];
                float o1 = x1 * c - x2 * s;
                float o2 = x2 * c + x1 * s;
                if (MODE == MODE_Q) { o1 *= 0.125f; o2 *= 0.125f; }
                size_t base = (MODE == MODE_Q)
                    ? ((size_t)bn * T_LEN + row) * DH                 // q[head][t][d]
                    : ((size_t)bn * S_ALL + SINKN + row) * DH;        // kall[kv][64+t][d]
                outb[base + d]      = f2bf(o1);
                outb[base + d + 32] = f2bf(o2);
            }
        }
    }
}

// ---------------- flash attention ----------------
// grid: (qtile 0..31, head 0..31) ; 128 threads = 4 waves ; wave w -> 16 query rows
__global__ __launch_bounds__(128)
void attn_kernel(const bfu* __restrict__ qb,   // [NH][T][DH] bf16 (rope'd, scaled)
                 const bfu* __restrict__ kall, // [NKV][S_ALL][DH]
                 const bfu* __restrict__ vall, // [NKV][S_ALL][DH]
                 bfu* __restrict__ ctx) {      // [T][D_MODEL]
    __shared__ bfu Qs[64 * 72];      // [t][d]
    __shared__ bfu Ks[64 * 72];      // [s][d]
    __shared__ bfu Vt[64 * 72];      // [d][s] (transposed)
    __shared__ bfu Ps[4][16 * 72];   // per-wave P [t][s]

    const int tid  = threadIdx.x;
    const int wave = tid >> 5, lane = tid & 31;
    const int half = lane >> 4, l16 = lane & 15;
    const int qt = blockIdx.x, h = blockIdx.y;
    const int kvh = h >> 2;

    const int lrow = tid >> 1, lseg = tid & 1; // Q/K staging: 64B each (async)
    const int va = tid >> 2;                   // V staging: s-pair base 2*va
    const int vd = (tid & 3) * 16;             // V staging: d segment

    // async-load Q tile (64x64); completion folded into first key-tile wait
    async_copy64(qb + ((size_t)h * T_LEN + qt * 64 + lrow) * DH + lseg * 32,
                 Qs + lrow * 72 + lseg * 32);

    v8f o[4];
#pragma unroll
    for (int i = 0; i < 4; ++i)
#pragma unroll
        for (int j = 0; j < 8; ++j) o[i][j] = 0.0f;
    float m_i[8], l_i[8];
#pragma unroll
    for (int r = 0; r < 8; ++r) { m_i[r] = -3.0e38f; l_i[r] = 0.0f; }

    for (int s0 = 0; s0 < S_ALL; s0 += 64) {
        __syncthreads(); // previous iteration finished reading Ks/Vt
        // stage K tile [s][d] : async global->LDS
        async_copy64(kall + ((size_t)kvh * S_ALL + s0 + lrow) * DH + lseg * 32,
                     Ks + lrow * 72 + lseg * 32);
        { // stage V tile transposed [d][s] : 2 s-rows x 16 d, packed b32 stores
            const bfu* gv0 = vall + ((size_t)kvh * S_ALL + s0 + 2 * va) * DH + vd;
            union { uint4 q[2]; bfu h[16]; } r0, r1;
            r0.q[0] = ((const uint4*)gv0)[0];
            r0.q[1] = ((const uint4*)gv0)[1];
            r1.q[0] = ((const uint4*)(gv0 + DH))[0];
            r1.q[1] = ((const uint4*)(gv0 + DH))[1];
#pragma unroll
            for (int j = 0; j < 16; ++j) {
                unsigned pk = (unsigned)r0.h[j] | ((unsigned)r1.h[j] << 16);
                *(unsigned*)(Vt + (vd + j) * 72 + 2 * va) = pk;
            }
        }
        wait_async0();
        __syncthreads();

        // ---- scores S = Q Kt : M=16 rows, N=64 keys, K=DH(2x32) ----
        v8f sc[4];
#pragma unroll
        for (int i = 0; i < 4; ++i)
#pragma unroll
            for (int j = 0; j < 8; ++j) sc[i][j] = 0.0f;
#pragma unroll
        for (int kk = 0; kk < 2; ++kk) {
            FragU a;
            const bfu* ap = Qs + (wave * 16 + l16) * 72 + kk * 32;
            a.u4[0] = *(const uint4*)(ap + 16 * half);
            a.u4[1] = *(const uint4*)(ap + 16 * half + 8);
#pragma unroll
            for (int nt = 0; nt < 4; ++nt) {
                FragU b; // B[d][s] = Ks[s][d] -> lane = key, contiguous d
                const bfu* bp = Ks + (nt * 16 + l16) * 72 + kk * 32 + 16 * half;
                b.u4[0] = ((const uint4*)bp)[0];
                b.u4[1] = ((const uint4*)bp)[1];
                sc[nt] = __builtin_amdgcn_wmma_f32_16x16x32_bf16(
                    false, a.v, false, b.v, (short)0, sc[nt], false, false);
            }
        }

        // ---- online softmax (row = r + 8*half, spread over 16 lanes/half) ----
        float p[4][8];
#pragma unroll
        for (int r = 0; r < 8; ++r) {
            float mx = fmaxf(fmaxf(sc[0][r], sc[1][r]), fmaxf(sc[2][r], sc[3][r]));
            mx = fmaxf(mx, __shfl_xor(mx, 1, 32));
            mx = fmaxf(mx, __shfl_xor(mx, 2, 32));
            mx = fmaxf(mx, __shfl_xor(mx, 4, 32));
            mx = fmaxf(mx, __shfl_xor(mx, 8, 32));
            float mnew = fmaxf(m_i[r], mx);
            float corr = __expf(m_i[r] - mnew);
            float rs = 0.0f;
#pragma unroll
            for (int nt = 0; nt < 4; ++nt) { p[nt][r] = __expf(sc[nt][r] - mnew); rs += p[nt][r]; }
            rs += __shfl_xor(rs, 1, 32);
            rs += __shfl_xor(rs, 2, 32);
            rs += __shfl_xor(rs, 4, 32);
            rs += __shfl_xor(rs, 8, 32);
            l_i[r] = l_i[r] * corr + rs;
            m_i[r] = mnew;
#pragma unroll
            for (int dt = 0; dt < 4; ++dt) o[dt][r] *= corr;
        }

        // ---- C-layout -> A-layout via per-wave LDS ----
#pragma unroll
        for (int nt = 0; nt < 4; ++nt)
#pragma unroll
            for (int r = 0; r < 8; ++r)
                Ps[wave][(r + 8 * half) * 72 + nt * 16 + l16] = f2bf(p[nt][r]);
        __syncthreads();

        // ---- O += P V : M=16, N=DH, K=64 keys (2x32) ----
#pragma unroll
        for (int kk = 0; kk < 2; ++kk) {
            FragU a;
            const bfu* ap = &Ps[wave][l16 * 72 + kk * 32];
            a.u4[0] = *(const uint4*)(ap + 16 * half);
            a.u4[1] = *(const uint4*)(ap + 16 * half + 8);
#pragma unroll
            for (int dt = 0; dt < 4; ++dt) {
                FragU b; // B[s][d] = Vt[d][s] -> lane = d, contiguous s
                const bfu* bp = Vt + (dt * 16 + l16) * 72 + kk * 32 + 16 * half;
                b.u4[0] = ((const uint4*)bp)[0];
                b.u4[1] = ((const uint4*)bp)[1];
                o[dt] = __builtin_amdgcn_wmma_f32_16x16x32_bf16(
                    false, a.v, false, b.v, (short)0, o[dt], false, false);
            }
        }
    }

    // ---- normalize, write ctx[t][h*64+d] as bf16 (A of the Wo GEMM) ----
    float invl[8];
#pragma unroll
    for (int r = 0; r < 8; ++r) invl[r] = 1.0f / l_i[r];
#pragma unroll
    for (int dt = 0; dt < 4; ++dt) {
        int d = dt * 16 + l16;
#pragma unroll
        for (int r = 0; r < 8; ++r) {
            int row = qt * 64 + wave * 16 + r + 8 * half;
            ctx[(size_t)row * D_MODEL + h * DH + d] = f2bf(o[dt][r] * invl[r]);
        }
    }
}

// ---------------- host launcher ----------------
extern "C" void kernel_launch(void* const* d_in, const int* in_sizes, int n_in,
                              void* d_out, int out_size, void* d_ws, size_t ws_size,
                              hipStream_t stream) {
    (void)in_sizes; (void)n_in; (void)out_size; (void)ws_size;
    const float* x   = (const float*)d_in[0];
    const float* kvc = (const float*)d_in[1];
    const float* Wq  = (const float*)d_in[2];
    const float* Wk  = (const float*)d_in[3];
    const float* Wv  = (const float*)d_in[4];
    const float* Wo  = (const float*)d_in[5];
    const float* bo  = (const float*)d_in[6];
    float* out = (float*)d_out;

    char* ws = (char*)d_ws;
    size_t off = 0;
    auto carve = [&](size_t bytes) { size_t p = off; off += (bytes + 255) & ~(size_t)255; return p; };
    bfu* xb   = (bfu*)(ws + carve(sizeof(bfu) * (size_t)T_LEN * D_MODEL));
    bfu* qb   = (bfu*)(ws + carve(sizeof(bfu) * (size_t)NH * T_LEN * DH));
    bfu* kall = (bfu*)(ws + carve(sizeof(bfu) * (size_t)NKV * S_ALL * DH));
    bfu* vall = (bfu*)(ws + carve(sizeof(bfu) * (size_t)NKV * S_ALL * DH));
    bfu* ctx  = (bfu*)(ws + carve(sizeof(bfu) * (size_t)T_LEN * D_MODEL));

    cvt_f32_bf16<<<(T_LEN * D_MODEL / 4) / 256, 256, 0, stream>>>(x, xb);
    sink_copy<<<(NKV * SINKN * DH / 4) / 256, 256, 0, stream>>>(kvc, kall, vall);

    gemm_bf16<MODE_Q><<<dim3(T_LEN / 128, NH),  256, 0, stream>>>(xb, Wq, qb,   nullptr, nullptr, NH * DH);
    gemm_bf16<MODE_K><<<dim3(T_LEN / 128, NKV), 256, 0, stream>>>(xb, Wk, kall, nullptr, nullptr, NKV * DH);
    gemm_bf16<MODE_V><<<dim3(T_LEN / 128, NKV), 256, 0, stream>>>(xb, Wv, vall, nullptr, nullptr, NKV * DH);

    attn_kernel<<<dim3(T_LEN / 64, NH), 128, 0, stream>>>(qb, kall, vall, ctx);

    gemm_bf16<MODE_O><<<dim3(T_LEN / 128, D_MODEL / 64), 256, 0, stream>>>(ctx, Wo, nullptr, out, bo, D_MODEL);
}
